// XCA_30133490549472
// MI455X (gfx1250) — matrix operand order repaired and checked by reading, not compiled, not run on hardware
//
#include <hip/hip_runtime.h>
#include <stdint.h>

// Problem constants: B=8, N=4096, C=768, H=8, D=96, 3C=2304
#define NB 8
#define NN 4096
#define NC 768
#define NH 8
#define ND 96
#define NC3 2304

typedef __bf16 bf16_t;
typedef bf16_t v16bf __attribute__((ext_vector_type(16)));
typedef float  v8f   __attribute__((ext_vector_type(8)));

union Frag {
  v16bf v;
  uint4 u[2];
};

__device__ __forceinline__ unsigned short f2bf(float f) {
  unsigned int u = __float_as_uint(f);
  u += 0x7FFFu + ((u >> 16) & 1u);   // round-to-nearest-even
  return (unsigned short)(u >> 16);
}

__device__ __forceinline__ v8f wmma_bf16(Frag a, Frag b, v8f c) {
  // D = A(16x32 bf16) x B(32x16 bf16) + C(16x16 f32)
  return __builtin_amdgcn_wmma_f32_16x16x32_bf16(false, a.v, false, b.v,
                                                 (short)0, c, false, false);
}

// ---- CDNA5 async global->LDS copy (ASYNCcnt-tracked, no VGPR round trip) ----
// Low 32 bits of a generic pointer to __shared__ data are the LDS byte address
// (flat-address truncation rule, ISA 10.2).
__device__ __forceinline__ uint32_t lds_off(const void* p) {
  return (uint32_t)(size_t)p;
}
__device__ __forceinline__ void async_copy_b128(uint32_t ldsaddr, const void* gptr) {
  asm volatile("global_load_async_to_lds_b128 %0, %1, off"
               :: "v"(ldsaddr), "v"((unsigned long long)(size_t)gptr)
               : "memory");
}
__device__ __forceinline__ void wait_async_le4() {
  asm volatile("s_wait_asynccnt 0x4" ::: "memory");
}
__device__ __forceinline__ void wait_async_0() {
  asm volatile("s_wait_asynccnt 0x0" ::: "memory");
}

// ---------------------------------------------------------------- conversions
__global__ __launch_bounds__(256) void conv_x_bf16(const float* __restrict__ in,
                                                   unsigned short* __restrict__ out,
                                                   int n4) {
  int idx = blockIdx.x * 256 + threadIdx.x;
  if (idx >= n4) return;
  float4 f = *(const float4*)(in + (size_t)idx * 4);
  unsigned short* o = out + (size_t)idx * 4;
  o[0] = f2bf(f.x); o[1] = f2bf(f.y); o[2] = f2bf(f.z); o[3] = f2bf(f.w);
}

// in[R][Cc] f32  ->  out[Cc][R] bf16 (transpose)
__global__ __launch_bounds__(256) void convT_bf16(const float* __restrict__ in,
                                                  unsigned short* __restrict__ out,
                                                  int R, int Cc) {
  int tid = blockIdx.x * 256 + threadIdx.x;
  if (tid >= R * Cc) return;
  int c = tid / R;
  int r = tid - c * R;
  out[tid] = f2bf(in[(size_t)r * Cc + c]);
}

__global__ __launch_bounds__(256) void zero_f32(float* __restrict__ p, int n) {
  int idx = blockIdx.x * 256 + threadIdx.x;
  if (idx < n) p[idx] = 0.f;
}

// ------------------------------------------------- GEMM1: QKV = X @ Wqkv,
// epilogue scatters into Q[b,h,d,n], K[b,h,e,n], V[b,h,n,e] (bf16)
// 128x128 block tile, 8 waves (4x2), K-step 32, double-buffered LDS fed by
// async global->LDS copies.
__global__ __launch_bounds__(256) void qkv_gemm_scatter(
    const unsigned short* __restrict__ Xh,      // [32768][768]
    const unsigned short* __restrict__ WqkvT,   // [2304][768]
    unsigned short* __restrict__ Qs,            // [64][96][4096]
    unsigned short* __restrict__ Ks,            // [64][96][4096]
    unsigned short* __restrict__ Vs) {          // [64][4096][96]
  __shared__ __align__(16) unsigned short Alds[2][128 * 40];
  __shared__ __align__(16) unsigned short Blds[2][128 * 40];
  const int tid  = threadIdx.x;
  const int lane = tid & 31;
  const int w    = tid >> 5;
  const int wm   = w & 3;        // 4 waves along M (32 rows each)
  const int wn   = w >> 2;       // 2 waves along N (64 cols each)
  const int l    = lane & 15;
  const int hi   = lane >> 4;
  const int gm0  = blockIdx.x * 128;
  const int gn0  = blockIdx.y * 128;

  // loader: 512 uint4 chunks per operand tile; 2 per thread
  const int r0 = tid >> 1, q0 = (tid & 1) * 8;            // unused (loader uses ra, qa below)
  const int c1 = tid + 256;
  const int r1 = c1 >> 2, q1 = (c1 & 3) * 8;
  // NOTE: chunk c=tid maps to row=tid>>2, qo=(tid&3)*8:
  const int ra = tid >> 2, qa = (tid & 3) * 8;

  const unsigned short* Ag0 = Xh    + (size_t)(gm0 + ra) * NC + qa;
  const unsigned short* Ag1 = Xh    + (size_t)(gm0 + r1) * NC + q1;
  const unsigned short* Bg0 = WqkvT + (size_t)(gn0 + ra) * NC + qa;
  const unsigned short* Bg1 = WqkvT + (size_t)(gn0 + r1) * NC + q1;

  uint32_t LA0[2], LA1[2], LB0[2], LB1[2];
#pragma unroll
  for (int b = 0; b < 2; ++b) {
    LA0[b] = lds_off(&Alds[b][ra * 40 + qa]);
    LA1[b] = lds_off(&Alds[b][r1 * 40 + q1]);
    LB0[b] = lds_off(&Blds[b][ra * 40 + qa]);
    LB1[b] = lds_off(&Blds[b][r1 * 40 + q1]);
  }
  (void)r0; (void)q0;

  v8f acc[2][4];
#pragma unroll
  for (int i = 0; i < 2; ++i)
#pragma unroll
    for (int j = 0; j < 4; ++j)
      acc[i][j] = (v8f){0.f, 0.f, 0.f, 0.f, 0.f, 0.f, 0.f, 0.f};

  // prologue: fill buffer 0
  async_copy_b128(LA0[0], Ag0);
  async_copy_b128(LA1[0], Ag1);
  async_copy_b128(LB0[0], Bg0);
  async_copy_b128(LB1[0], Bg1);

  for (int k0 = 0, it = 0; k0 < NC; k0 += 32, ++it) {
    const int cur = it & 1;
    const int nxt = cur ^ 1;
    if (k0 + 32 < NC) {
      async_copy_b128(LA0[nxt], Ag0 + k0 + 32);
      async_copy_b128(LA1[nxt], Ag1 + k0 + 32);
      async_copy_b128(LB0[nxt], Bg0 + k0 + 32);
      async_copy_b128(LB1[nxt], Bg1 + k0 + 32);
      wait_async_le4();          // current buffer's 4 copies complete
    } else {
      wait_async_0();
    }
    __syncthreads();

    Frag a[2], bfr[4];
#pragma unroll
    for (int mi = 0; mi < 2; ++mi) {
      const unsigned short* p = &Alds[cur][(wm * 32 + mi * 16 + l) * 40 + hi * 8];
      a[mi].u[0] = *(const uint4*)(p);
      a[mi].u[1] = *(const uint4*)(p + 16);
    }
#pragma unroll
    for (int ni = 0; ni < 4; ++ni) {
      const unsigned short* p = &Blds[cur][(wn * 64 + ni * 16 + l) * 40 + hi * 8];
      bfr[ni].u[0] = *(const uint4*)(p);
      bfr[ni].u[1] = *(const uint4*)(p + 16);
    }
#pragma unroll
    for (int mi = 0; mi < 2; ++mi)
#pragma unroll
      for (int ni = 0; ni < 4; ++ni)
        acc[mi][ni] = wmma_bf16(a[mi], bfr[ni], acc[mi][ni]);
    __syncthreads();             // all waves done with `cur` before it is refilled
  }

  // epilogue: scatter into q/k/v layouts
#pragma unroll
  for (int mi = 0; mi < 2; ++mi)
#pragma unroll
    for (int ni = 0; ni < 4; ++ni)
#pragma unroll
      for (int i = 0; i < 8; ++i) {
        const int row = gm0 + wm * 32 + mi * 16 + hi * 8 + i;  // b*N + n
        const int col = gn0 + wn * 64 + ni * 16 + l;           // c3
        const int bb = row >> 12, n = row & 4095;
        const int h = col / 288;
        const int r = col - h * 288;
        const int d = r / 3;
        const int s = r - d * 3;
        const int bh = bb * NH + h;
        const unsigned short val = f2bf(acc[mi][ni][i]);
        if (s == 0)      Qs[((size_t)bh * ND + d) * NN + n] = val;
        else if (s == 1) Ks[((size_t)bh * ND + d) * NN + n] = val;
        else             Vs[((size_t)bh * NN + n) * ND + d] = val;
      }
}

// ---------------------------------------------- per-(b,h,d) inverse L2 norms
__global__ __launch_bounds__(256) void colnorm(const unsigned short* __restrict__ Qs,
                                               const unsigned short* __restrict__ Ks,
                                               float* __restrict__ rq,
                                               float* __restrict__ rk) {
  const int id   = blockIdx.x;               // 0..12287
  const int which = (id >= NB * NH * ND) ? 1 : 0;
  const int bhd  = which ? id - NB * NH * ND : id;
  const unsigned short* src = (which ? Ks : Qs) + (size_t)bhd * NN;
  const int t = threadIdx.x;

  float s = 0.f;
#pragma unroll
  for (int pass = 0; pass < 2; ++pass) {
    uint4 u = *(const uint4*)(src + pass * 2048 + t * 8);
    unsigned int ww[4] = {u.x, u.y, u.z, u.w};
#pragma unroll
    for (int j = 0; j < 4; ++j) {
      float lo = __uint_as_float(ww[j] << 16);
      float hh = __uint_as_float(ww[j] & 0xFFFF0000u);
      s += lo * lo + hh * hh;
    }
  }
#pragma unroll
  for (int off = 16; off > 0; off >>= 1) s += __shfl_down(s, off, 32);

  __shared__ float wsum[8];
  if ((t & 31) == 0) wsum[t >> 5] = s;
  __syncthreads();
  if (t == 0) {
    float tot = 0.f;
#pragma unroll
    for (int i = 0; i < 8; ++i) tot += wsum[i];
    (which ? rk : rq)[bhd] = rsqrtf(tot);
  }
}

// --------------------------------- Gram: S[bh][d][e] += sum_n q[d,n]*k[e,n]
// split-K over n (8 slices of 512), f32 atomics into zeroed Sbuf
__global__ __launch_bounds__(128) void gram_gemm(const unsigned short* __restrict__ Qs,
                                                 const unsigned short* __restrict__ Ks,
                                                 float* __restrict__ Sbuf) {
  const int bh    = blockIdx.x;
  const int slice = blockIdx.y;
  const int tid   = threadIdx.x;
  const int lane  = tid & 31;
  const int w     = tid >> 5;
  const int wm    = w & 1;       // d half (48 rows)
  const int wn    = w >> 1;      // e half (48 cols)
  const int l     = lane & 15;
  const int hi    = lane >> 4;
  const unsigned short* Qb = Qs + (size_t)bh * ND * NN;
  const unsigned short* Kb = Ks + (size_t)bh * ND * NN;

  v8f acc[3][3];
#pragma unroll
  for (int i = 0; i < 3; ++i)
#pragma unroll
    for (int j = 0; j < 3; ++j)
      acc[i][j] = (v8f){0.f, 0.f, 0.f, 0.f, 0.f, 0.f, 0.f, 0.f};

  const int nbeg = slice * 512;
  for (int n0 = nbeg; n0 < nbeg + 512; n0 += 32) {
    Frag a[3], bfr[3];
#pragma unroll
    for (int mi = 0; mi < 3; ++mi) {
      const unsigned short* p = Qb + (size_t)(wm * 48 + mi * 16 + l) * NN + n0 + hi * 8;
      a[mi].u[0] = *(const uint4*)(p);
      a[mi].u[1] = *(const uint4*)(p + 16);
    }
#pragma unroll
    for (int ni = 0; ni < 3; ++ni) {
      const unsigned short* p = Kb + (size_t)(wn * 48 + ni * 16 + l) * NN + n0 + hi * 8;
      bfr[ni].u[0] = *(const uint4*)(p);
      bfr[ni].u[1] = *(const uint4*)(p + 16);
    }
#pragma unroll
    for (int mi = 0; mi < 3; ++mi)
#pragma unroll
      for (int ni = 0; ni < 3; ++ni)
        acc[mi][ni] = wmma_bf16(a[mi], bfr[ni], acc[mi][ni]);
  }

#pragma unroll
  for (int mi = 0; mi < 3; ++mi)
#pragma unroll
    for (int ni = 0; ni < 3; ++ni)
#pragma unroll
      for (int i = 0; i < 8; ++i) {
        const int d = wm * 48 + mi * 16 + hi * 8 + i;
        const int e = wn * 48 + ni * 16 + l;
        atomicAdd(&Sbuf[(size_t)bh * (ND * ND) + d * ND + e], acc[mi][ni][i]);
      }
}

// ---------------- softmax over e with normalization + temperature folded in
__global__ __launch_bounds__(128) void softmax_k(const float* __restrict__ Sbuf,
                                                 const float* __restrict__ rq,
                                                 const float* __restrict__ rk,
                                                 const float* __restrict__ temp,
                                                 unsigned short* __restrict__ attnh) {
  const int bhd = blockIdx.x;        // 0..6143
  const int bh  = bhd / ND;
  const int d   = bhd - bh * ND;
  const int h   = bh & (NH - 1);
  const int t   = threadIdx.x;

  float v = -INFINITY;
  const float sd = rq[bh * ND + d] * temp[h];
  if (t < ND)
    v = Sbuf[(size_t)bh * (ND * ND) + d * ND + t] * sd * rk[bh * ND + t];

  __shared__ float red[128];
  red[t] = v;
  __syncthreads();
#pragma unroll
  for (int s2 = 64; s2 > 0; s2 >>= 1) {
    if (t < s2) red[t] = fmaxf(red[t], red[t + s2]);
    __syncthreads();
  }
  const float m = red[0];
  __syncthreads();
  const float ex = (t < ND) ? expf(v - m) : 0.f;
  red[t] = ex;
  __syncthreads();
#pragma unroll
  for (int s2 = 64; s2 > 0; s2 >>= 1) {
    if (t < s2) red[t] += red[t + s2];
    __syncthreads();
  }
  if (t < ND)
    attnh[(size_t)bh * (ND * ND) + d * ND + t] = f2bf(ex / red[0]);
}

// -------------------- y[n][d] = sum_e V[n][e] * attn[d][e]  (per b,h)
__global__ __launch_bounds__(256) void apply_attn(const unsigned short* __restrict__ Vs,
                                                  const unsigned short* __restrict__ attnh,
                                                  unsigned short* __restrict__ Ymid) {
  const int bh   = blockIdx.x;
  const int bb   = bh >> 3;
  const int h    = bh & 7;
  const int tid  = threadIdx.x;
  const int lane = tid & 31;
  const int w    = tid >> 5;
  const int l    = lane & 15;
  const int hi   = lane >> 4;
  const int n0   = blockIdx.y * 256 + w * 32;
  const unsigned short* Vb = Vs + (size_t)bh * NN * ND;
  const unsigned short* Ab = attnh + (size_t)bh * (ND * ND);

  v8f acc[2][6];
#pragma unroll
  for (int i = 0; i < 2; ++i)
#pragma unroll
    for (int j = 0; j < 6; ++j)
      acc[i][j] = (v8f){0.f, 0.f, 0.f, 0.f, 0.f, 0.f, 0.f, 0.f};

#pragma unroll
  for (int e0 = 0; e0 < ND; e0 += 32) {
    Frag a[2], bfr[6];
#pragma unroll
    for (int mi = 0; mi < 2; ++mi) {
      const unsigned short* p = Vb + (size_t)(n0 + mi * 16 + l) * ND + e0 + hi * 8;
      a[mi].u[0] = *(const uint4*)(p);
      a[mi].u[1] = *(const uint4*)(p + 16);
    }
#pragma unroll
    for (int ni = 0; ni < 6; ++ni) {
      const unsigned short* p = Ab + (ni * 16 + l) * ND + e0 + hi * 8;
      bfr[ni].u[0] = *(const uint4*)(p);
      bfr[ni].u[1] = *(const uint4*)(p + 16);
    }
#pragma unroll
    for (int mi = 0; mi < 2; ++mi)
#pragma unroll
      for (int ni = 0; ni < 6; ++ni)
        acc[mi][ni] = wmma_bf16(a[mi], bfr[ni], acc[mi][ni]);
  }

#pragma unroll
  for (int mi = 0; mi < 2; ++mi)
#pragma unroll
    for (int ni = 0; ni < 6; ++ni)
#pragma unroll
      for (int i = 0; i < 8; ++i) {
        const int n = n0 + mi * 16 + hi * 8 + i;
        const int dcol = ni * 16 + l;
        Ymid[((size_t)(bb * NN + n)) * NC + h * ND + dcol] = f2bf(acc[mi][ni][i]);
      }
}

// -------------------------- final projection: out = Ymid @ Wproj + b_proj
__global__ __launch_bounds__(256) void proj_gemm(
    const unsigned short* __restrict__ Yh,      // [32768][768]
    const unsigned short* __restrict__ WprojT,  // [768][768]
    const float* __restrict__ bias,
    float* __restrict__ out) {
  __shared__ __align__(16) unsigned short Alds[2][128 * 40];
  __shared__ __align__(16) unsigned short Blds[2][128 * 40];
  const int tid  = threadIdx.x;
  const int lane = tid & 31;
  const int w    = tid >> 5;
  const int wm   = w & 3;
  const int wn   = w >> 2;
  const int l    = lane & 15;
  const int hi   = lane >> 4;
  const int gm0  = blockIdx.x * 128;
  const int gn0  = blockIdx.y * 128;

  const int c1 = tid + 256;
  const int r1 = c1 >> 2, q1 = (c1 & 3) * 8;
  const int ra = tid >> 2, qa = (tid & 3) * 8;

  const unsigned short* Ag0 = Yh     + (size_t)(gm0 + ra) * NC + qa;
  const unsigned short* Ag1 = Yh     + (size_t)(gm0 + r1) * NC + q1;
  const unsigned short* Bg0 = WprojT + (size_t)(gn0 + ra) * NC + qa;
  const unsigned short* Bg1 = WprojT + (size_t)(gn0 + r1) * NC + q1;

  uint32_t LA0[2], LA1[2], LB0[2], LB1[2];
#pragma unroll
  for (int b = 0; b < 2; ++b) {
    LA0[b] = lds_off(&Alds[b][ra * 40 + qa]);
    LA1[b] = lds_off(&Alds[b][r1 * 40 + q1]);
    LB0[b] = lds_off(&Blds[b][ra * 40 + qa]);
    LB1[b] = lds_off(&Blds[b][r1 * 40 + q1]);
  }

  v8f acc[2][4];
#pragma unroll
  for (int i = 0; i < 2; ++i)
#pragma unroll
    for (int j = 0; j < 4; ++j)
      acc[i][j] = (v8f){0.f, 0.f, 0.f, 0.f, 0.f, 0.f, 0.f, 0.f};

  async_copy_b128(LA0[0], Ag0);
  async_copy_b128(LA1[0], Ag1);
  async_copy_b128(LB0[0], Bg0);
  async_copy_b128(LB1[0], Bg1);

  for (int k0 = 0, it = 0; k0 < NC; k0 += 32, ++it) {
    const int cur = it & 1;
    const int nxt = cur ^ 1;
    if (k0 + 32 < NC) {
      async_copy_b128(LA0[nxt], Ag0 + k0 + 32);
      async_copy_b128(LA1[nxt], Ag1 + k0 + 32);
      async_copy_b128(LB0[nxt], Bg0 + k0 + 32);
      async_copy_b128(LB1[nxt], Bg1 + k0 + 32);
      wait_async_le4();
    } else {
      wait_async_0();
    }
    __syncthreads();

    Frag a[2], bfr[4];
#pragma unroll
    for (int mi = 0; mi < 2; ++mi) {
      const unsigned short* p = &Alds[cur][(wm * 32 + mi * 16 + l) * 40 + hi * 8];
      a[mi].u[0] = *(const uint4*)(p);
      a[mi].u[1] = *(const uint4*)(p + 16);
    }
#pragma unroll
    for (int ni = 0; ni < 4; ++ni) {
      const unsigned short* p = &Blds[cur][(wn * 64 + ni * 16 + l) * 40 + hi * 8];
      bfr[ni].u[0] = *(const uint4*)(p);
      bfr[ni].u[1] = *(const uint4*)(p + 16);
    }
#pragma unroll
    for (int mi = 0; mi < 2; ++mi)
#pragma unroll
      for (int ni = 0; ni < 4; ++ni)
        acc[mi][ni] = wmma_bf16(a[mi], bfr[ni], acc[mi][ni]);
    __syncthreads();
  }

#pragma unroll
  for (int mi = 0; mi < 2; ++mi)
#pragma unroll
    for (int ni = 0; ni < 4; ++ni)
#pragma unroll
      for (int i = 0; i < 8; ++i) {
        const int row = gm0 + wm * 32 + mi * 16 + hi * 8 + i;
        const int col = gn0 + wn * 64 + ni * 16 + l;
        out[(size_t)row * NC + col] = acc[mi][ni][i] + bias[col];
      }
}

// ------------------------------------------------------------------ launcher
extern "C" void kernel_launch(void* const* d_in, const int* in_sizes, int n_in,
                              void* d_out, int out_size, void* d_ws, size_t ws_size,
                              hipStream_t stream) {
  const float* x     = (const float*)d_in[0];   // [8,4096,768]
  const float* Wqkv  = (const float*)d_in[1];   // [768,2304]
  const float* Wproj = (const float*)d_in[2];   // [768,768]
  const float* bproj = (const float*)d_in[3];   // [768]
  const float* temp  = (const float*)d_in[4];   // [8]
  float* out = (float*)d_out;

  char* ws = (char*)d_ws;
  size_t off = 0;
  auto alloc = [&](size_t bytes) -> void* {
    void* p = ws + off;
    off += (bytes + 255) & ~(size_t)255;
    return p;
  };
  unsigned short* Xh     = (unsigned short*)alloc((size_t)NB * NN * NC * 2);      // 50.3 MB
  unsigned short* WqkvT  = (unsigned short*)alloc((size_t)NC3 * NC * 2);          // 3.5 MB
  unsigned short* WprojT = (unsigned short*)alloc((size_t)NC * NC * 2);           // 1.2 MB
  unsigned short* Qs     = (unsigned short*)alloc((size_t)NB * NH * ND * NN * 2); // 50.3 MB
  unsigned short* Ks     = (unsigned short*)alloc((size_t)NB * NH * ND * NN * 2); // 50.3 MB
  unsigned short* Vs     = (unsigned short*)alloc((size_t)NB * NH * NN * ND * 2); // 50.3 MB
  unsigned short* Ymid   = (unsigned short*)alloc((size_t)NB * NN * NC * 2);      // 50.3 MB
  float* Sbuf            = (float*)alloc((size_t)NB * NH * ND * ND * 4);          // 2.4 MB
  unsigned short* attnh  = (unsigned short*)alloc((size_t)NB * NH * ND * ND * 2); // 1.2 MB
  float* rq              = (float*)alloc((size_t)NB * NH * ND * 4);
  float* rk              = (float*)alloc((size_t)NB * NH * ND * 4);

  // 1) bf16 conversions (+ weight transposes so both WMMA operands are K-contiguous)
  conv_x_bf16<<<(NB * NN * NC / 4 + 255) / 256, 256, 0, stream>>>(x, Xh, NB * NN * NC / 4);
  convT_bf16<<<(NC * NC3 + 255) / 256, 256, 0, stream>>>(Wqkv, WqkvT, NC, NC3);
  convT_bf16<<<(NC * NC + 255) / 256, 256, 0, stream>>>(Wproj, WprojT, NC, NC);
  zero_f32<<<(NB * NH * ND * ND + 255) / 256, 256, 0, stream>>>(Sbuf, NB * NH * ND * ND);

  // 2) QKV GEMM with fused reshape/split scatter (async double-buffered)
  qkv_gemm_scatter<<<dim3((NB * NN) / 128, NC3 / 128), 256, 0, stream>>>(Xh, WqkvT, Qs, Ks, Vs);

  // 3) per-(b,h,d) inverse L2 norms over the patch axis
  colnorm<<<2 * NB * NH * ND, 256, 0, stream>>>(Qs, Ks, rq, rk);

  // 4) Gram matrix QtK (split-K=8, atomics into Sbuf)
  gram_gemm<<<dim3(NB * NH, 8), 128, 0, stream>>>(Qs, Ks, Sbuf);

  // 5) softmax with norms + temperature folded in
  softmax_k<<<NB * NH * ND, 128, 0, stream>>>(Sbuf, rq, rk, temp, attnh);

  // 6) apply attention to V, fold heads back into Ymid[b,n,c]
  apply_attn<<<dim3(NB * NH, NN / 256), 256, 0, stream>>>(Vs, attnh, Ymid);

  // 7) output projection + bias (f32 out, async double-buffered)
  proj_gemm<<<dim3((NB * NN) / 128, NC / 128), 256, 0, stream>>>(Ymid, WprojT, bproj, out);
}
